// SHMA_27393301414339
// MI455X (gfx1250) — matrix operand order repaired and checked by reading, not compiled, not run on hardware
//
#include <hip/hip_runtime.h>
#include <hip/hip_bf16.h>

// ---------------------------------------------------------------------------
// Windowed single-head attention + gated MLP, MI455X (gfx1250, wave32, WMMA).
// GEMMs on v_wmma_f32_16x16x32_bf16; shared B operands staged into LDS with
// gfx1250 async-to-LDS loads (ASYNCcnt) and read back as strided 16-bit LDS loads.
// ---------------------------------------------------------------------------

typedef __bf16 bf16;
typedef bf16 v16bf __attribute__((ext_vector_type(16)));
typedef bf16 v8bf  __attribute__((ext_vector_type(8)));
typedef float v8f  __attribute__((ext_vector_type(8)));

#define SCALE_C 0.125f   // DA^-0.5, DA=64

__device__ __forceinline__ v8f wmma_bf16(v16bf a, v16bf b, v8f c) {
  return __builtin_amdgcn_wmma_f32_16x16x32_bf16(
      /*neg_a=*/false, a, /*neg_b=*/false, b,
      /*c_mod=*/(short)0, c, /*reuse_a=*/false, /*reuse_b=*/false);
}

union FragU { v16bf v; struct { v8bf lo, hi; } p; };

// A fragment (16x32 bf16, row-major, ld elems). Lanes 0-15: rows M=0..15,
// K={0..7,16..23}; lanes 16-31: K={8..15,24..31}. Two 16B loads per lane.
__device__ __forceinline__ v16bf frag_a(const bf16* base, int ld) {
  const int lane = threadIdx.x & 31;
  const int half = lane >> 4;
  const int r    = lane & 15;
  const bf16* rp = base + (size_t)r * ld + half * 8;
  FragU f;
  f.p.lo = *(const v8bf*)(rp);
  f.p.hi = *(const v8bf*)(rp + 16);
  return f.v;
}

// B fragment (32x16 bf16, row-major B[K][N], ld elems). Lane l holds column
// N=l%16; lanes 0-15: K={0..7,16..23}; lanes 16-31: K={8..15,24..31}.
__device__ __forceinline__ v16bf frag_b(const bf16* base, int ld) {
  const int lane = threadIdx.x & 31;
  const int half = lane >> 4;
  const bf16* cp = base + (size_t)(half * 8) * ld + (lane & 15);
  v16bf v;
#pragma unroll
  for (int i = 0; i < 8; ++i) {
    v[i]     = cp[(size_t)i * ld];
    v[8 + i] = cp[(size_t)(16 + i) * ld];
  }
  return v;
}

__device__ __forceinline__ float sigmoidf_dev(float x) {
  return 1.0f / (1.0f + __expf(-x));
}

// gfx1250 async copy: 16B global -> LDS, tracked by ASYNCcnt.
__device__ __forceinline__ void async_g2l_b128(unsigned lds_off,
                                               const bf16* gptr) {
  unsigned long long ga = (unsigned long long)(size_t)gptr;
  asm volatile("global_load_async_to_lds_b128 %0, %1, off"
               :: "v"(lds_off), "v"(ga) : "memory");
}
__device__ __forceinline__ void wait_asynccnt0() {
  asm volatile("s_wait_asynccnt 0x0" ::: "memory");
}

// ---------------------------------------------------------------------------
// Kernel 1: window-partition + fp32->bf16 pack: x[B,C,H,W] -> xf[Bw][C][N]
// float4 loads / 4x bf16 (8B) stores, coalesced on both sides.
// ---------------------------------------------------------------------------
__global__ void __launch_bounds__(256) k_pack_x(const float* __restrict__ x,
                                                bf16* __restrict__ xf) {
  size_t t = (size_t)blockIdx.x * 256 + threadIdx.x;   // 4,194,304 threads
  int n4 = (int)(t & 63) * 4;                          // n = n4..n4+3
  int c  = (int)((t >> 6) & 255);
  int w  = (int)(t >> 14);
  int b  = w >> 6, wh = (w >> 3) & 7, ww = w & 7;
  int ty = n4 >> 4, tx = n4 & 15;
  const float4 v = *(const float4*)&x[
      (((size_t)(b * 256 + c) * 128) + wh * 16 + ty) * 128 + ww * 16 + tx];
  bf16* d = xf + t * 4;
  d[0] = (bf16)v.x; d[1] = (bf16)v.y; d[2] = (bf16)v.z; d[3] = (bf16)v.w;
}

// fp32 -> bf16 flat convert (weight packing)
__global__ void __launch_bounds__(256) k_cvt(const float* __restrict__ s,
                                             bf16* __restrict__ d, int n) {
  int t = blockIdx.x * 256 + threadIdx.x;
  if (t < n) d[t] = (bf16)s[t];
}

#define XROW 264   // padded LDS row (bf16 elems): lanes hit disjoint bank quads

// ---------------------------------------------------------------------------
// Kernel 2: q = (q_w @ xf)*s+b scaled, k = (k_w @ xf)*s+b.  [64,256] each.
// xf window (128KB) staged once into LDS via async loads; all 8 waves then
// stream B fragments from LDS. Wave-uniform base+stride epilogue (no per-
// element selects): addr = base + n*sN + da*sD.
// ---------------------------------------------------------------------------
__global__ void __launch_bounds__(256) k_qk(const bf16* __restrict__ xf,
                                            const bf16* __restrict__ qw,
                                            const bf16* __restrict__ kw,
                                            const float* __restrict__ qs,
                                            const float* __restrict__ qb,
                                            const float* __restrict__ ks,
                                            const float* __restrict__ kb,
                                            bf16* __restrict__ qT,
                                            bf16* __restrict__ kk) {
  __shared__ __attribute__((aligned(16))) bf16 xsl[256 * XROW];  // 135,168 B
  const int w    = blockIdx.x;
  const int wave = threadIdx.x >> 5;
  const int lane = threadIdx.x & 31;
  const int half = lane >> 4, lc = lane & 15;
  const bf16* x = xf + (size_t)w * 65536;

  // --- async stage xf[256][256] -> xsl[256][XROW] ---
  {
    const unsigned lbase = (unsigned)(size_t)(&xsl[0]);
    const int n0 = lane * 8;
#pragma unroll 4
    for (int it = 0; it < 32; ++it) {
      int c = it * 8 + wave;
      async_g2l_b128(lbase + (unsigned)(c * (XROW * 2) + n0 * 2),
                     x + (size_t)c * 256 + n0);
    }
    wait_asynccnt0();
  }
  __syncthreads();

  const bool isK = wave >= 4;
  const int  mt  = wave & 3;
  const bf16*  Wm = isK ? kw : qw;
  const float* Sv = isK ? ks : qs;
  const float* Bv = isK ? kb : qb;
  const float  sc = isK ? 1.0f : SCALE_C;
  // unified destination addressing (selects hoisted out of all loops):
  //   q: qT[w][n][da]  -> sN=64,  sD=1
  //   k: kk[w][da][n]  -> sN=1,   sD=256
  bf16* dbase  = isK ? (kk + (size_t)w * 64 * 256) : (qT + (size_t)w * 256 * 64);
  const int sN = isK ? 1 : 64;
  const int sD = isK ? 256 : 1;

  v16bf a[8];
#pragma unroll
  for (int k0 = 0; k0 < 8; ++k0)
    a[k0] = frag_a(Wm + (size_t)(mt * 16) * 256 + k0 * 32, 256);

  for (int nt = 0; nt < 16; ++nt) {
    v16bf bfr[8];
#pragma unroll
    for (int k0 = 0; k0 < 8; ++k0)
      bfr[k0] = frag_b(&xsl[(k0 * 32) * XROW + nt * 16], XROW);
    v8f acc = {};
#pragma unroll
    for (int k0 = 0; k0 < 8; ++k0)
      acc = wmma_bf16(a[k0], bfr[k0], acc);

    const int n   = nt * 16 + lc;
    const int da0 = mt * 16 + 8 * half;
    bf16* dp = dbase + (size_t)n * sN + (size_t)da0 * sD;
#pragma unroll
    for (int v = 0; v < 8; ++v) {
      int da  = da0 + v;
      float t = (acc[v] * Sv[da] + Bv[da]) * sc;
      dp[(size_t)v * sD] = (bf16)t;
    }
  }
}

// ---------------------------------------------------------------------------
// Kernel 3: S = qT @ k (256x256, K=64) -> LDS f32 -> row softmax -> attnT bf16.
// k[64][256] async-staged to LDS (shared by all 8 waves). LDS rows padded
// (261 f32 / 264 bf16) for conflict-free strided access.
// ---------------------------------------------------------------------------
#define SROW 261
__global__ void __launch_bounds__(256) k_attn(const bf16* __restrict__ qT,
                                              const bf16* __restrict__ kk,
                                              bf16* __restrict__ attnT) {
  __shared__ __attribute__((aligned(16))) float Sl[256 * SROW];   // 267,264 B
  __shared__ __attribute__((aligned(16))) bf16  ksl[64 * XROW];   //  33,792 B
  const int w    = blockIdx.x;
  const int wave = threadIdx.x >> 5;
  const int lane = threadIdx.x & 31;
  const int half = lane >> 4, lc = lane & 15;
  const bf16* q = qT + (size_t)w * 256 * 64;
  const bf16* k = kk + (size_t)w * 64 * 256;

  // --- async stage k[64][256] -> ksl[64][XROW] ---
  {
    const unsigned lbase = (unsigned)(size_t)(&ksl[0]);
    const int n0 = lane * 8;
#pragma unroll
    for (int it = 0; it < 8; ++it) {
      int c = it * 8 + wave;
      async_g2l_b128(lbase + (unsigned)(c * (XROW * 2) + n0 * 2),
                     k + (size_t)c * 256 + n0);
    }
    wait_asynccnt0();
  }
  __syncthreads();

  for (int mi = 0; mi < 2; ++mi) {
    const int mt = wave * 2 + mi;
    v16bf a0 = frag_a(q + (size_t)(mt * 16) * 64 + 0, 64);
    v16bf a1 = frag_a(q + (size_t)(mt * 16) * 64 + 32, 64);
    for (int nt = 0; nt < 16; ++nt) {
      v16bf b0 = frag_b(&ksl[nt * 16], XROW);
      v16bf b1 = frag_b(&ksl[32 * XROW + nt * 16], XROW);
      v8f acc = {};
      acc = wmma_bf16(a0, b0, acc);
      acc = wmma_bf16(a1, b1, acc);
#pragma unroll
      for (int v = 0; v < 8; ++v)
        Sl[(mt * 16 + v + 8 * half) * SROW + nt * 16 + lc] = acc[v];
    }
  }
  __syncthreads();

  // softmax: thread t owns attention row n=t (reduce over m)
  {
    const int t = threadIdx.x;
    float* row = &Sl[t * SROW];
    float m = -1e30f;
    for (int j = 0; j < 256; ++j) m = fmaxf(m, row[j]);
    float s = 0.f;
    for (int j = 0; j < 256; ++j) { float e = __expf(row[j] - m); row[j] = e; s += e; }
    float inv = 1.f / s;
    bf16* at = attnT + (size_t)w * 65536;
    for (int j = 0; j < 256; ++j)            // column t of attnT: coalesced in t
      at[(size_t)j * 256 + t] = (bf16)(row[j] * inv);
  }
}

// ---------------------------------------------------------------------------
// Kernel 4: per (window, 64-row mid slab):
//   v,gate = sigmoid(vg_w @ xf * s + b)  (xf async-staged in LDS)
//   mid    = (v @ attnT) * gate          (v fragments from LDS)
// ---------------------------------------------------------------------------
#define VROW 264
__global__ void __launch_bounds__(256) k_va(const bf16* __restrict__ xf,
                                            const bf16* __restrict__ vgw,
                                            const float* __restrict__ vgs,
                                            const float* __restrict__ vgb,
                                            const bf16* __restrict__ attnT,
                                            bf16* __restrict__ mid) {
  __shared__ __attribute__((aligned(16))) bf16 xsl[256 * XROW];   // 135,168 B
  __shared__ __attribute__((aligned(16))) bf16 vsl[64 * VROW];    //  33,792 B
  __shared__ __attribute__((aligned(16))) bf16 gsl[64 * VROW];    //  33,792 B
  const int w    = blockIdx.x;          // window
  const int cb   = blockIdx.y;          // mid slab: rows cb*64 .. +63
  const int wave = threadIdx.x >> 5;
  const int lane = threadIdx.x & 31;
  const int half = lane >> 4, lc = lane & 15;
  const bf16* x = xf + (size_t)w * 65536;

  // --- async stage xf window -> LDS ---
  {
    const unsigned lbase = (unsigned)(size_t)(&xsl[0]);
    const int n0 = lane * 8;
#pragma unroll 4
    for (int it = 0; it < 32; ++it) {
      int c = it * 8 + wave;
      async_g2l_b128(lbase + (unsigned)(c * (XROW * 2) + n0 * 2),
                     x + (size_t)c * 256 + n0);
    }
    wait_asynccnt0();
  }
  __syncthreads();

  // --- step 1: v / gate slab GEMMs (+affine+sigmoid) into LDS ---
  {
    const int  mt    = wave;                       // 0..7
    const bool isG   = mt >= 4;
    const int  rbase = (isG ? 1024 : 0) + cb * 64 + (mt & 3) * 16;
    const int  rl    = (mt & 3) * 16;
    bf16* dst = isG ? gsl : vsl;
    v16bf a[8];
#pragma unroll
    for (int k0 = 0; k0 < 8; ++k0)
      a[k0] = frag_a(vgw + (size_t)rbase * 256 + k0 * 32, 256);
    for (int nt = 0; nt < 16; ++nt) {
      v16bf bfr[8];
#pragma unroll
      for (int k0 = 0; k0 < 8; ++k0)
        bfr[k0] = frag_b(&xsl[(k0 * 32) * XROW + nt * 16], XROW);
      v8f acc = {};
#pragma unroll
      for (int k0 = 0; k0 < 8; ++k0)
        acc = wmma_bf16(a[k0], bfr[k0], acc);
#pragma unroll
      for (int v = 0; v < 8; ++v) {
        int gr = rbase + v + 8 * half;
        float t = sigmoidf_dev(acc[v] * vgs[gr] + vgb[gr]);
        dst[(rl + v + 8 * half) * VROW + nt * 16 + lc] = (bf16)t;
      }
    }
  }
  __syncthreads();

  // --- step 2: (v @ attnT) * gate -> mid ---
  {
    const int mt = wave & 3;            // m-tile within slab
    const int nh = wave >> 2;           // n half: tiles nh*8 .. nh*8+7
    const bf16* at = attnT + (size_t)w * 65536;
    v16bf a[8];
#pragma unroll
    for (int k0 = 0; k0 < 8; ++k0)
      a[k0] = frag_a(&vsl[(mt * 16) * VROW + k0 * 32], VROW);
    for (int ni = 0; ni < 8; ++ni) {
      const int nt = nh * 8 + ni;
      v16bf bfr[8];
#pragma unroll
      for (int k0 = 0; k0 < 8; ++k0)
        bfr[k0] = frag_b(at + (size_t)(k0 * 32) * 256 + nt * 16, 256);
      v8f acc = {};
#pragma unroll
      for (int k0 = 0; k0 < 8; ++k0)
        acc = wmma_bf16(a[k0], bfr[k0], acc);
#pragma unroll
      for (int v = 0; v < 8; ++v) {
        int rl = mt * 16 + v + 8 * half;
        int n  = nt * 16 + lc;
        float g = (float)gsl[rl * VROW + n];
        mid[(((size_t)w * 1024) + cb * 64 + rl) * 256 + n] = (bf16)(acc[v] * g);
      }
    }
  }
}

// ---------------------------------------------------------------------------
// Kernel 5: out = p_w @ mid * s + b, fused window_reverse -> d_out f32.
// [256 x 256] per window, K=1024. 4-wide n-blocking shares the A fragment;
// B fragments batched per k-step (single clause, overlapped latency).
// ---------------------------------------------------------------------------
__global__ void __launch_bounds__(256) k_proj(const bf16* __restrict__ mid,
                                              const bf16* __restrict__ pw,
                                              const float* __restrict__ ps,
                                              const float* __restrict__ pb,
                                              float* __restrict__ out) {
  const int w    = blockIdx.x;
  const int b    = w >> 6, wh = (w >> 3) & 7, ww = w & 7;
  const int wave = threadIdx.x >> 5;
  const int lane = threadIdx.x & 31;
  const int half = lane >> 4, lc = lane & 15;
  const bf16* m = mid + (size_t)w * 1024 * 256;

  for (int mi = 0; mi < 2; ++mi) {
    const int mt = wave * 2 + mi;
    for (int ng = 0; ng < 4; ++ng) {
      v8f acc[4] = {};
      for (int kc = 0; kc < 4; ++kc) {
        v16bf a[8];
#pragma unroll
        for (int t = 0; t < 8; ++t)
          a[t] = frag_a(pw + (size_t)(mt * 16) * 1024 + (kc * 8 + t) * 32, 1024);
#pragma unroll
        for (int t = 0; t < 8; ++t) {
          const int k0 = kc * 8 + t;
          v16bf bj[4];
#pragma unroll
          for (int j = 0; j < 4; ++j)
            bj[j] = frag_b(m + (size_t)(k0 * 32) * 256 + (ng * 4 + j) * 16, 256);
#pragma unroll
          for (int j = 0; j < 4; ++j)
            acc[j] = wmma_bf16(a[t], bj[j], acc[j]);
        }
      }
#pragma unroll
      for (int j = 0; j < 4; ++j) {
#pragma unroll
        for (int v = 0; v < 8; ++v) {
          int c  = mt * 16 + v + 8 * half;
          int ty = ng * 4 + j;               // n = (ng*4+j)*16 + lc
          float t = acc[j][v] * ps[c] + pb[c];
          out[(((size_t)(b * 256 + c) * 128) + wh * 16 + ty) * 128 + ww * 16 + lc] = t;
        }
      }
    }
  }
}

// ---------------------------------------------------------------------------
// Host-side launch. Workspace layout (bytes):
//   xf @0 (33.5M) | qT @33,554,432 (8.4M) | k @41,943,040 (8.4M)
//   attnT @50,331,648 (33.5M) | mid @83,886,080 (134M)
//   qw @218,103,808 | kw +32K | vgw +64K | pw +64K+1M
// ---------------------------------------------------------------------------
extern "C" void kernel_launch(void* const* d_in, const int* in_sizes, int n_in,
                              void* d_out, int out_size, void* d_ws, size_t ws_size,
                              hipStream_t stream) {
  (void)in_sizes; (void)n_in; (void)out_size; (void)ws_size;
  const float* x    = (const float*)d_in[0];
  const float* q_w  = (const float*)d_in[1];
  const float* q_s  = (const float*)d_in[2];
  const float* q_b  = (const float*)d_in[3];
  const float* k_w  = (const float*)d_in[4];
  const float* k_s  = (const float*)d_in[5];
  const float* k_b  = (const float*)d_in[6];
  const float* vg_w = (const float*)d_in[7];
  const float* vg_s = (const float*)d_in[8];
  const float* vg_b = (const float*)d_in[9];
  const float* p_w  = (const float*)d_in[10];
  const float* p_s  = (const float*)d_in[11];
  const float* p_b  = (const float*)d_in[12];
  float* out = (float*)d_out;

  char* ws = (char*)d_ws;
  bf16* xf     = (bf16*)(ws + 0);
  bf16* qT     = (bf16*)(ws + 33554432ull);
  bf16* kk     = (bf16*)(ws + 41943040ull);
  bf16* attnT  = (bf16*)(ws + 50331648ull);
  bf16* mid    = (bf16*)(ws + 83886080ull);
  bf16* qw_bf  = (bf16*)(ws + 218103808ull);
  bf16* kw_bf  = (bf16*)(ws + 218103808ull + 32768ull);
  bf16* vgw_bf = (bf16*)(ws + 218103808ull + 65536ull);
  bf16* pw_bf  = (bf16*)(ws + 218103808ull + 65536ull + 1048576ull);

  // 1) pack activations and weights to bf16
  k_pack_x<<<16384, 256, 0, stream>>>(x, xf);
  k_cvt<<<64, 256, 0, stream>>>(q_w, qw_bf, 16384);
  k_cvt<<<64, 256, 0, stream>>>(k_w, kw_bf, 16384);
  k_cvt<<<2048, 256, 0, stream>>>(vg_w, vgw_bf, 524288);
  k_cvt<<<1024, 256, 0, stream>>>(p_w, pw_bf, 262144);

  // 2) q / k projections
  k_qk<<<256, 256, 0, stream>>>(xf, qw_bf, kw_bf, q_s, q_b, k_s, k_b, qT, kk);

  // 3) attention scores + softmax
  k_attn<<<256, 256, 0, stream>>>(qT, kk, attnT);

  // 4) v/gate projection + attention apply + gating
  k_va<<<dim3(256, 16), 256, 0, stream>>>(xf, vgw_bf, vg_s, vg_b, attnT, mid);

  // 5) output projection + window reverse
  k_proj<<<256, 256, 0, stream>>>(mid, pw_bf, p_s, p_b, out);
}
